// GDTransformer_33663953666494
// MI455X (gfx1250) — matrix operand order repaired and checked by reading, not compiled, not run on hardware
//
#include <hip/hip_runtime.h>

#define DD 1024
#define SS 1024
#define BB 2
#define VV 32000
#define NLAYERS 4

typedef __attribute__((ext_vector_type(16))) __bf16 v16bf;
typedef __attribute__((ext_vector_type(8)))  __bf16 bf16x8;
typedef __attribute__((ext_vector_type(8)))  float  v8f;
typedef __attribute__((ext_vector_type(4)))  unsigned int u32x4;
typedef __attribute__((ext_vector_type(8)))  int i32x8;
typedef __attribute__((ext_vector_type(4)))  int i32x4;

#if defined(__has_builtin)
#if __has_builtin(__builtin_amdgcn_tensor_load_to_lds) && __has_builtin(__builtin_amdgcn_s_wait_tensorcnt)
#define HAVE_TDM 1
#endif
#endif

__device__ __forceinline__ v8f wmma_bf16(v16bf a, v16bf b, v8f c) {
  return __builtin_amdgcn_wmma_f32_16x16x32_bf16(false, a, false, b, (short)0, c, false, false);
}

// ---------------------------------------------------------------------------
// small utility kernels
// ---------------------------------------------------------------------------
__global__ void cvt_f32_bf16(const float* __restrict__ in, __bf16* __restrict__ out, long n) {
  long i = (long)blockIdx.x * 256 + threadIdx.x;
  if (i < n) out[i] = (__bf16)in[i];
}

__global__ void zero_f32(float* __restrict__ p, long n) {
  long i = (long)blockIdx.x * 256 + threadIdx.x;
  if (i < n) p[i] = 0.0f;
}

__global__ __launch_bounds__(256) void transpose_bf16(const __bf16* __restrict__ in,
                                                      __bf16* __restrict__ out, int R, int C) {
  __shared__ __bf16 t[32][33];
  int c0 = blockIdx.x * 32, r0 = blockIdx.y * 32;
  int tx = threadIdx.x & 31, ty = threadIdx.x >> 5;
  for (int i = ty; i < 32; i += 8) t[i][tx] = in[(size_t)(r0 + i) * C + c0 + tx];
  __syncthreads();
  for (int i = ty; i < 32; i += 8) out[(size_t)(c0 + i) * R + r0 + tx] = t[tx][i];
}

// eT[b][d][s] = TE[x[b,s]][d]  (f32, transposed so flash epilogue reads contiguous s)
__global__ void gather_eT(const int* __restrict__ x, const float* __restrict__ TE,
                          float* __restrict__ eT) {
  int bs = blockIdx.x;
  int b = bs >> 10, s = bs & 1023;
  int row = x[bs];
  for (int d = threadIdx.x; d < DD; d += 256)
    eT[((size_t)b * DD + d) * SS + s] = TE[(size_t)row * DD + d];
}

// ftb[b][s][d] = bf16(f[b][s][d]) for s in [0,S)
__global__ void build_ftb(const float* __restrict__ f, __bf16* __restrict__ out) {
  long i = (long)blockIdx.x * 256 + threadIdx.x;
  if (i < (long)BB * SS * DD) {
    long b = i / ((long)SS * DD);
    out[i] = (__bf16)f[i + b * DD];
  }
}

// ---------------------------------------------------------------------------
// generic NT GEMM:  C[m][n] = sum_k A[m][k] * Bm[n][k]    (bf16 in, f32 acc)
// MODE 0: store bf16, MODE 1: store f32, MODE 2: += f32
// tile: BM=BN=128, BK=32, 8 waves as 2x4, each wave 64x32 (4x2 WMMA tiles)
// ---------------------------------------------------------------------------
template <int MODE>
__global__ __launch_bounds__(256) void gemm_nt(const __bf16* __restrict__ A,
                                               const __bf16* __restrict__ Bm,
                                               void* __restrict__ Cp,
                                               int lda, int ldb, int ldc, int K,
                                               long strideA, long strideB, long strideC) {
  __shared__ __align__(32) __bf16 As[128 * 32];
  __shared__ __align__(32) __bf16 Bs[128 * 32];
  const int z = blockIdx.z;
  A  += (size_t)z * strideA;
  Bm += (size_t)z * strideB;
  const int wave = threadIdx.x >> 5, lane = threadIdx.x & 31;
  const int wm = wave >> 2, wn = wave & 3;
  const int h = lane >> 4, nl = lane & 15;
  const int m0 = blockIdx.x * 128, n0 = blockIdx.y * 128;
  v8f acc[4][2] = {};
  const int lr = threadIdx.x >> 1, lc = (threadIdx.x & 1) * 16;
  for (int kb = 0; kb < K; kb += 32) {
    *(v16bf*)&As[lr * 32 + lc] = *(const v16bf*)&A[(size_t)(m0 + lr) * lda + kb + lc];
    *(v16bf*)&Bs[lr * 32 + lc] = *(const v16bf*)&Bm[(size_t)(n0 + lr) * ldb + kb + lc];
    __syncthreads();
    v16bf bfrag[2];
#pragma unroll
    for (int j = 0; j < 2; ++j)
      bfrag[j] = *(const v16bf*)&Bs[(wn * 32 + j * 16 + nl) * 32 + 16 * h];
#pragma unroll
    for (int i = 0; i < 4; ++i) {
      union { v16bf v; bf16x8 q[2]; } ua;
      const int ar = wm * 64 + i * 16 + nl;
      ua.v = *(const v16bf*)&As[ar * 32 + 16 * h];
#pragma unroll
      for (int j = 0; j < 2; ++j) acc[i][j] = wmma_bf16(ua.v, bfrag[j], acc[i][j]);
    }
    __syncthreads();
  }
#pragma unroll
  for (int i = 0; i < 4; ++i) {
#pragma unroll
    for (int j = 0; j < 2; ++j) {
      const int tmb = m0 + wm * 64 + i * 16;
      const int col = n0 + wn * 32 + j * 16 + nl;
#pragma unroll
      for (int e = 0; e < 8; ++e) {
        const size_t idx = (size_t)z * strideC + (size_t)(tmb + e + 8 * h) * ldc + col;
        if (MODE == 0)      ((__bf16*)Cp)[idx] = (__bf16)acc[i][j][e];
        else if (MODE == 1) ((float*)Cp)[idx] = acc[i][j][e];
        else                ((float*)Cp)[idx] += acc[i][j][e];
      }
    }
  }
}

// ---------------------------------------------------------------------------
// causal softmax over scores row, scale 1/sqrt(D); write bf16 attn
// ---------------------------------------------------------------------------
__global__ __launch_bounds__(256) void causal_softmax(const float* __restrict__ sc,
                                                      __bf16* __restrict__ attn) {
  const int l = blockIdx.x, t = threadIdx.x;
  const float scale = 0.03125f;  // 1/32
  __shared__ float red[256];
  float mx = -1e30f;
  for (int s = t; s <= l; s += 256) mx = fmaxf(mx, sc[(size_t)l * SS + s] * scale);
  red[t] = mx; __syncthreads();
  for (int o = 128; o; o >>= 1) { if (t < o) red[t] = fmaxf(red[t], red[t + o]); __syncthreads(); }
  mx = red[0]; __syncthreads();
  float sum = 0.0f;
  for (int s = t; s <= l; s += 256) sum += __expf(sc[(size_t)l * SS + s] * scale - mx);
  red[t] = sum; __syncthreads();
  for (int o = 128; o; o >>= 1) { if (t < o) red[t] += red[t + o]; __syncthreads(); }
  const float inv = 1.0f / red[0];
  for (int s = t; s < SS; s += 256) {
    float v = (s <= l) ? __expf(sc[(size_t)l * SS + s] * scale - mx) * inv : 0.0f;
    attn[(size_t)l * SS + s] = (__bf16)v;
  }
}

// ---------------------------------------------------------------------------
// fused flash-over-vocab: one 16-row tile per WG, 8 waves.
// Q tile loaded via TDM (tensor_load_to_lds, TENSORcnt) when available.
// phase1: logits tile (16x128) vs TEb;  phase2: Ew += P @ TE via TEbT.
// epilogue: errT[b][d][s] = bf16(e - Ew)
// ---------------------------------------------------------------------------
__global__ __launch_bounds__(256) void flash_vocab(const __bf16* __restrict__ ftb,
                                                   const __bf16* __restrict__ TEb,
                                                   const __bf16* __restrict__ TEbT,
                                                   const float* __restrict__ eT,
                                                   __bf16* __restrict__ errT) {
  __shared__ __align__(32) __bf16 Qs[16 * DD];
  __shared__ __align__(32) __bf16 Ps[16 * 128];
  __shared__ float redm[128], reds[128];
  __shared__ float mrow[16], crow[16], lrow[16];

  const int wave = threadIdx.x >> 5, lane = threadIdx.x & 31;
  const int h = lane >> 4, nl = lane & 15;
  const int r0 = blockIdx.x * 16;
  const int b = r0 >> 10, s0 = r0 & 1023;

#if defined(HAVE_TDM)
  // ---- Tensor Data Mover: one DMA for the whole 32KB Q tile (16 x 2048B, 1-D)
  if (threadIdx.x < 32) {
    unsigned long long ga = (unsigned long long)(const void*)&ftb[(size_t)r0 * DD];
    // generic pointer to LDS: low 32 bits are the LDS byte address (ISA 10.2)
    unsigned ldsa = (unsigned)(unsigned long long)(const void*)&Qs[0];
    // D# group0: count=1 | lds_addr | global_addr[56:0] | type=2 (bits 127:126)
    u32x4 g0;
    g0.x = 1u;
    g0.y = ldsa;
    g0.z = (unsigned)ga;
    g0.w = (unsigned)((ga >> 32) & 0x01FFFFFFull) | 0x80000000u;
    // D# group1: data_size=3 (8B units); tensor_dim0 = tile_dim0 = 4096 units = 32KB
    i32x8 g1;
    g1[0] = 0x00030000;  // workgroup_mask=0, data_size=3
    g1[1] = 0x10000000;  // tensor_dim0[15:0]=0x1000 in bits 63:48
    g1[2] = 0x00010000;  // tensor_dim0 hi=0, tensor_dim1=1 (bits 111:80 low part)
    g1[3] = 0x10000000;  // tile_dim0=0x1000 in bits 127:112
    g1[4] = 0;           // tile_dim1=0 (unused), tile_dim2=0
    g1[5] = 0x00001000;  // tensor_dim0_stride = 4096 (benign for 1-D)
    g1[6] = 0;
    g1[7] = 0;
    i32x4 gz; gz[0] = 0; gz[1] = 0; gz[2] = 0; gz[3] = 0;
    i32x8 gz8; gz8[0] = 0; gz8[1] = 0; gz8[2] = 0; gz8[3] = 0;
    gz8[4] = 0; gz8[5] = 0; gz8[6] = 0; gz8[7] = 0;
    __builtin_amdgcn_tensor_load_to_lds(g0, g1, gz, gz, gz8, 0);
    __builtin_amdgcn_s_wait_tensorcnt(0);
  }
#else
  for (int c = threadIdx.x; c < (16 * DD) / 16; c += 256)
    *(v16bf*)&Qs[c * 16] = *(const v16bf*)&ftb[(size_t)r0 * DD + (size_t)c * 16];
#endif
  if (threadIdx.x < 16) { mrow[threadIdx.x] = -1e30f; lrow[threadIdx.x] = 0.0f; }
  v8f acc[8] = {};
  __syncthreads();

  for (int vb = 0; vb < VV; vb += 128) {
    // ---- phase 1: 16x16 logits tile per wave (cols vb+16w .. +16)
    v8f sc = {};
    const __bf16* Bb = TEb + ((size_t)(vb + wave * 16 + nl) * DD + 16 * h);
    if (vb + 128 < VV) __builtin_prefetch(Bb + (size_t)128 * DD, 0, 1);
#pragma unroll 4
    for (int kb = 0; kb < DD; kb += 32) {
      union { v16bf v; bf16x8 q[2]; } ua;
      ua.v = *(const v16bf*)&Qs[nl * DD + kb + 16 * h];
      v16bf bfr = *(const v16bf*)(Bb + kb);
      sc = wmma_bf16(ua.v, bfr, sc);
    }
    // ---- per-row block max (reduce over 16 lanes in each half)
#pragma unroll
    for (int e = 0; e < 8; ++e) {
      float v = sc[e];
      v = fmaxf(v, __shfl_xor(v, 1));
      v = fmaxf(v, __shfl_xor(v, 2));
      v = fmaxf(v, __shfl_xor(v, 4));
      v = fmaxf(v, __shfl_xor(v, 8));
      if (nl == 0) redm[wave * 16 + e + 8 * h] = v;
    }
    __syncthreads();
    if (threadIdx.x < 16) {
      const int rr = threadIdx.x;
      float bm = redm[rr];
#pragma unroll
      for (int w = 1; w < 8; ++w) bm = fmaxf(bm, redm[w * 16 + rr]);
      const float mo = mrow[rr];
      const float mn = fmaxf(mo, bm);
      const float cc = __expf(mo - mn);
      mrow[rr] = mn; crow[rr] = cc; lrow[rr] *= cc;
    }
    __syncthreads();
    // ---- rescale accumulators, P = exp(sc - m), partial row sums
#pragma unroll
    for (int e = 0; e < 8; ++e) {
      const int rr = e + 8 * h;
      const float cc = crow[rr];
      const float mn = mrow[rr];
#pragma unroll
      for (int t = 0; t < 8; ++t) acc[t][e] *= cc;
      const float p = __expf(sc[e] - mn);
      Ps[rr * 128 + wave * 16 + nl] = (__bf16)p;
      float v = p;
      v += __shfl_xor(v, 1);
      v += __shfl_xor(v, 2);
      v += __shfl_xor(v, 4);
      v += __shfl_xor(v, 8);
      if (nl == 0) reds[wave * 16 + rr] = v;
    }
    __syncthreads();
    if (threadIdx.x < 16) {
      float s = 0.0f;
#pragma unroll
      for (int w = 0; w < 8; ++w) s += reds[w * 16 + threadIdx.x];
      lrow[threadIdx.x] += s;
    }
    // ---- phase 2: acc[nt] += P(16x128) @ TEblk(128 x 16) via TEbT (contiguous)
#pragma unroll
    for (int kt = 0; kt < 4; ++kt) {
      union { v16bf v; bf16x8 q[2]; } ua;
      ua.v = *(const v16bf*)&Ps[nl * 128 + kt * 32 + 16 * h];
#pragma unroll
      for (int nt = 0; nt < 8; ++nt) {
        const __bf16* Bt = TEbT + ((size_t)(wave * 128 + nt * 16 + nl) * VV + vb + kt * 32 + 16 * h);
        v16bf bfr = *(const v16bf*)Bt;
        acc[nt] = wmma_bf16(ua.v, bfr, acc[nt]);
      }
    }
    __syncthreads();
  }

  // ---- epilogue: err = e - acc/l, write transposed bf16
  float rinv[8];
#pragma unroll
  for (int e = 0; e < 8; ++e) rinv[e] = 1.0f / (lrow[e + 8 * h] + 1e-8f);
  const int sbase = s0 + 8 * h;
#pragma unroll
  for (int nt = 0; nt < 8; ++nt) {
    const int d = wave * 128 + nt * 16 + nl;
    const float* ev = &eT[((size_t)b * DD + d) * SS + sbase];
    bf16x8 o;
#pragma unroll
    for (int e = 0; e < 8; ++e) o[e] = (__bf16)(ev[e] - acc[nt][e] * rinv[e]);
    *(bf16x8*)&errT[((size_t)b * DD + d) * SS + sbase] = o;
  }
}

// ---------------------------------------------------------------------------
// layernorm of last row of f per batch
// ---------------------------------------------------------------------------
__global__ __launch_bounds__(256) void ln_last(const float* __restrict__ f,
                                               const float* __restrict__ g,
                                               const float* __restrict__ be,
                                               float* __restrict__ fn) {
  const int b = blockIdx.x, t = threadIdx.x;
  const float* row = f + (size_t)b * (SS + 1) * DD + (size_t)SS * DD;
  __shared__ float red[256];
  float s = 0.0f;
  for (int d = t; d < DD; d += 256) s += row[d];
  red[t] = s; __syncthreads();
  for (int o = 128; o; o >>= 1) { if (t < o) red[t] += red[t + o]; __syncthreads(); }
  const float mu = red[0] / (float)DD; __syncthreads();
  float v = 0.0f;
  for (int d = t; d < DD; d += 256) { float x = row[d] - mu; v += x * x; }
  red[t] = v; __syncthreads();
  for (int o = 128; o; o >>= 1) { if (t < o) red[t] += red[t + o]; __syncthreads(); }
  const float rstd = rsqrtf(red[0] / (float)DD + 1e-5f);
  for (int d = t; d < DD; d += 256)
    fn[(size_t)b * DD + d] = (row[d] - mu) * rstd * g[d] + be[d];
}

// out[b][v] = fn[b] . TEb[v]   (one wave per v, lane-split over D)
__global__ __launch_bounds__(256) void out_logits(const float* __restrict__ fn,
                                                  const __bf16* __restrict__ TEb,
                                                  float* __restrict__ out) {
  const int wave = threadIdx.x >> 5, lane = threadIdx.x & 31;
  const long wg = (long)blockIdx.x * 8 + wave;  // 0 .. B*V-1
  const int b = (int)(wg / VV), v = (int)(wg % VV);
  const __bf16* te = TEb + (size_t)v * DD + lane * 32;
  const float* fb = fn + (size_t)b * DD + lane * 32;
  float acc = 0.0f;
#pragma unroll
  for (int i = 0; i < 32; i += 8) {
    bf16x8 tv = *(const bf16x8*)(te + i);
#pragma unroll
    for (int j = 0; j < 8; ++j) acc += fb[i + j] * (float)tv[j];
  }
  acc += __shfl_xor(acc, 1);
  acc += __shfl_xor(acc, 2);
  acc += __shfl_xor(acc, 4);
  acc += __shfl_xor(acc, 8);
  acc += __shfl_xor(acc, 16);
  if (lane == 0) out[wg] = acc;
}

// ---------------------------------------------------------------------------
extern "C" void kernel_launch(void* const* d_in, const int* in_sizes, int n_in,
                              void* d_out, int out_size, void* d_ws, size_t ws_size,
                              hipStream_t stream) {
  (void)in_sizes; (void)n_in; (void)out_size; (void)ws_size;
  const int*   x   = (const int*)d_in[0];
  const float* TE  = (const float*)d_in[1];
  const float* PE  = (const float*)d_in[2];
  const float* Wq  = (const float*)d_in[3];
  const float* Wk  = (const float*)d_in[4];
  const float* lng = (const float*)d_in[5];
  const float* lnb = (const float*)d_in[6];
  float* out = (float*)d_out;

  char* base = (char*)d_ws;
  size_t off = 0;
  auto take = [&](size_t nbytes) -> void* {
    void* p = base + off;
    off = (off + nbytes + 255) & ~(size_t)255;
    return p;
  };
  // ~173 MB workspace total
  __bf16* TEb    = (__bf16*)take((size_t)VV * DD * 2);
  __bf16* TEbT   = (__bf16*)take((size_t)DD * VV * 2);
  __bf16* pb     = (__bf16*)take((size_t)(SS + 1) * DD * 2);
  __bf16* Wqb    = (__bf16*)take((size_t)DD * DD * 2);
  __bf16* Wkb    = (__bf16*)take((size_t)DD * DD * 2);
  __bf16* keysb  = (__bf16*)take((size_t)SS * DD * 2);
  __bf16* qb     = (__bf16*)take((size_t)SS * DD * 2);
  float*  scores = (float*)take((size_t)SS * SS * 4);
  __bf16* attnb  = (__bf16*)take((size_t)SS * SS * 2);
  float*  eT     = (float*)take((size_t)BB * DD * SS * 4);
  float*  f      = (float*)take((size_t)BB * (SS + 1) * DD * 4);
  __bf16* ftb    = (__bf16*)take((size_t)BB * SS * DD * 2);
  __bf16* errT   = (__bf16*)take((size_t)BB * DD * SS * 2);
  float*  fn     = (float*)take((size_t)BB * DD * 4);

  // precompute bf16 operands
  cvt_f32_bf16<<<dim3(((long)VV * DD + 255) / 256), 256, 0, stream>>>(TE, TEb, (long)VV * DD);
  transpose_bf16<<<dim3(DD / 32, VV / 32), 256, 0, stream>>>(TEb, TEbT, VV, DD);
  cvt_f32_bf16<<<dim3(((SS + 1) * DD + 255) / 256), 256, 0, stream>>>(PE, pb, (long)(SS + 1) * DD);
  cvt_f32_bf16<<<dim3((DD * DD + 255) / 256), 256, 0, stream>>>(Wq, Wqb, (long)DD * DD);
  cvt_f32_bf16<<<dim3((DD * DD + 255) / 256), 256, 0, stream>>>(Wk, Wkb, (long)DD * DD);
  gather_eT<<<dim3(BB * SS), 256, 0, stream>>>(x, TE, eT);
  zero_f32<<<dim3((BB * (SS + 1) * DD + 255) / 256), 256, 0, stream>>>(f, (long)BB * (SS + 1) * DD);

  // keys = p[:S] @ Wk^T ; queries = p[1:S+1] @ Wq^T   (NT WMMA GEMMs)
  gemm_nt<0><<<dim3(8, 8, 1), 256, 0, stream>>>(pb,      Wkb, keysb, DD, DD, DD, DD, 0, 0, 0);
  gemm_nt<0><<<dim3(8, 8, 1), 256, 0, stream>>>(pb + DD, Wqb, qb,    DD, DD, DD, DD, 0, 0, 0);
  // scores = q @ k^T  (f32 out), then causal softmax -> attn (bf16)
  gemm_nt<1><<<dim3(8, 8, 1), 256, 0, stream>>>(qb, keysb, scores, DD, DD, SS, DD, 0, 0, 0);
  causal_softmax<<<dim3(SS), 256, 0, stream>>>(scores, attnb);

  for (int layer = 0; layer < NLAYERS; ++layer) {
    build_ftb<<<dim3(((long)BB * SS * DD + 255) / 256), 256, 0, stream>>>(f, ftb);
    flash_vocab<<<dim3(BB * SS / 16), 256, 0, stream>>>(ftb, TEb, TEbT, eT, errT);
    // f[b,1:] += attn @ err   (NT: B = errT, batched over z)
    gemm_nt<2><<<dim3(8, 8, BB), 256, 0, stream>>>(attnb, errT, f + DD,
        SS, SS, DD, SS, 0, (long)DD * SS, (long)(SS + 1) * DD);
  }

  ln_last<<<dim3(BB), 256, 0, stream>>>(f, lng, lnb, fn);
  out_logits<<<dim3(BB * VV / 8), 256, 0, stream>>>(fn, TEb, out);
}
